// KGAT_Encoder_58884001628283
// MI455X (gfx1250) — matrix operand order repaired and checked by reading, not compiled, not run on hardware
//
#include <hip/hip_runtime.h>
#include <math.h>

#define N_USERS  20000
#define N_ENT    80000
#define NNODES   (N_USERS + N_ENT)   // 100000
#define DIM      64
#define NEDGE    1250000
#define EPS_ATT  1e-9f
#define EPS_NRM  1e-12f

typedef __attribute__((ext_vector_type(16))) _Float16 v16h;
typedef __attribute__((ext_vector_type(8)))  float    v8f;

__device__ __forceinline__ float lrelu(float x) { return x >= 0.0f ? x : 0.01f * x; }

// ---- concat user/entity embeddings into ego, seed running sum ----------------
__global__ __launch_bounds__(256) void k_concat(const float* __restrict__ ue,
                                                const float* __restrict__ ee,
                                                float* __restrict__ ego,
                                                float* __restrict__ sum) {
    int i = blockIdx.x * 256 + threadIdx.x;
    if (i >= NNODES * DIM) return;
    float v = (i < N_USERS * DIM) ? ue[i] : ee[i - N_USERS * DIM];
    ego[i] = v;
    sum[i] = v;
}

// ---- per-layer node-state init ----------------------------------------------
__global__ __launch_bounds__(256) void k_init_nodes(float* __restrict__ aggr,
                                                    float* __restrict__ mx,
                                                    float* __restrict__ denom) {
    int i = blockIdx.x * 256 + threadIdx.x;
    if (i < NNODES * DIM) aggr[i] = 0.0f;
    if (i < NNODES) { mx[i] = -__builtin_huge_valf(); denom[i] = 0.0f; }
}

// ---- pass 1: logits[e] = lrelu(<h,t,r>), segment max over src ---------------
// one wave32 per edge, float2 per lane => coalesced 128B pairs, L2-resident
__global__ __launch_bounds__(256) void k_edge_logits(const float* __restrict__ ego,
                                                     const float* __restrict__ rel_emb,
                                                     const int* __restrict__ src,
                                                     const int* __restrict__ dst,
                                                     const int* __restrict__ rel,
                                                     float* __restrict__ logits,
                                                     float* __restrict__ mx) {
    int e    = blockIdx.x * (blockDim.x >> 5) + (threadIdx.x >> 5);
    int lane = threadIdx.x & 31;
    if (e >= NEDGE) return;
    int s = src[e], d = dst[e], r = rel[e];
    float2 h = ((const float2*)(ego     + (size_t)s * DIM))[lane];
    float2 t = ((const float2*)(ego     + (size_t)d * DIM))[lane];
    float2 g = ((const float2*)(rel_emb + (size_t)r * DIM))[lane];
    float part = h.x * t.x * g.x + h.y * t.y * g.y;
#pragma unroll
    for (int off = 16; off >= 1; off >>= 1) part += __shfl_xor(part, off, 32);
    if (lane == 0) {
        float lg = lrelu(part);
        logits[e] = lg;
        atomicMax(&mx[s], lg);     // global_atomic_max_num_f32
    }
}

// ---- pass 2: ex = exp(logit - mx[src]); segment sum -> denom ----------------
__global__ __launch_bounds__(256) void k_edge_exp(const int* __restrict__ src,
                                                  const float* __restrict__ mx,
                                                  float* __restrict__ logits,
                                                  float* __restrict__ denom) {
    int e = blockIdx.x * 256 + threadIdx.x;
    if (e >= NEDGE) return;
    int s = src[e];
    float ex = __expf(logits[e] - mx[s]);
    logits[e] = ex;
    atomicAdd(&denom[s], ex);
}

// ---- pass 3: aggr[src] += t * att -------------------------------------------
__global__ __launch_bounds__(256) void k_edge_aggr(const float* __restrict__ ego,
                                                   const int* __restrict__ src,
                                                   const int* __restrict__ dst,
                                                   const float* __restrict__ logits,
                                                   const float* __restrict__ denom,
                                                   float* __restrict__ aggr) {
    int e    = blockIdx.x * (blockDim.x >> 5) + (threadIdx.x >> 5);
    int lane = threadIdx.x & 31;
    if (e >= NEDGE) return;
    int s = src[e], d = dst[e];
    float att = logits[e] / (denom[s] + EPS_ATT);
    float2 t = ((const float2*)(ego + (size_t)d * DIM))[lane];
    float* ap = aggr + (size_t)s * DIM + lane * 2;
    atomicAdd(ap,     t.x * att);
    atomicAdd(ap + 1, t.y * att);
}

// ---- node update: new = lrelu((ego+aggr)@W1 + (ego*aggr)@W2); L2-normalize --
// 256 threads = 8 waves; 32 nodes/block; wave -> one 16x16 output tile,
// K=64 -> 2 k-steps x 2 weight matrices = 4 v_wmma_f32_16x16x32_f16 per wave.
// W1/W2 are staged in LDS PRE-SWIZZLED into B-fragment order so each fragment
// is one aligned 32B read (2x ds_load_b128) instead of 16 ds_load_u16.
__global__ __launch_bounds__(256) void k_layer_update(const float* __restrict__ ego,
                                                      const float* __restrict__ aggr,
                                                      const float* __restrict__ W1,
                                                      const float* __restrict__ W2,
                                                      float* __restrict__ ego_out,
                                                      float* __restrict__ sum) {
    // B-fragment layout: frag f = kt*4 + ct; within a frag, lane-major, 16
    // contiguous halfs per lane:  sWf[(f*32 + lane)*16 + j]
    __shared__ __align__(32) _Float16 sW1f[DIM * DIM];  // 8 KB
    __shared__ __align__(32) _Float16 sW2f[DIM * DIM];  // 8 KB
    __shared__ __align__(32) _Float16 sX1[32 * DIM];    // 4 KB  ego+aggr  (row-major)
    __shared__ __align__(32) _Float16 sX2[32 * DIM];    // 4 KB  ego*aggr
    __shared__ float sO[32 * DIM];                      // 8 KB
    __shared__ float sR[32];

    int tid = threadIdx.x;
    int nb  = blockIdx.x * 32;                // 3125 blocks * 32 = 100000 exactly

    // stage W1/W2 -> f16, swizzled to B-fragment order.
    // B layout (K x N tile, kb = hi*16 + j + kt*32, n = ct*16 + m, lane = hi*16+m)
    for (int i = tid; i < DIM * DIM; i += 256) {
        int k = i >> 6, n = i & 63;
        int kt = k >> 5, r = k & 31;
        int hi = r >> 4, j = r & 15;
        int ct = n >> 4, m = n & 15;
        int dst = (((kt * 4 + ct) * 32) + hi * 16 + m) * 16 + j;
        sW1f[dst] = (_Float16)W1[i];
        sW2f[dst] = (_Float16)W2[i];
    }
    // stage X1 = ego+aggr, X2 = ego*aggr (row-major; A-side chunks vectorize)
    for (int i = tid; i < 32 * DIM; i += 256) {
        size_t g = (size_t)nb * DIM + i;
        float e = ego[g], a = aggr[g];
        sX1[i] = (_Float16)(e + a);
        sX2[i] = (_Float16)(e * a);
    }
    __syncthreads();

    int wv = tid >> 5, lane = tid & 31;
    int rg = wv >> 2;                 // row group (0/1) -> nodes rg*16..rg*16+15
    int ct = wv & 3;                  // col tile  (0..3) -> out cols ct*16..
    int m  = lane & 15;
    int hi = (lane & 16) ? 1 : 0;

    v8f acc = {0.f, 0.f, 0.f, 0.f, 0.f, 0.f, 0.f, 0.f};
#pragma unroll
    for (int kt = 0; kt < 2; ++kt) {
        // A 16x32 f16 layout (matches B staging): element j holds K=hi*16+j;
        // M = lane%16.  For a fixed lane this is one contiguous 16-half run.
        v16h a1, a2;
#pragma unroll
        for (int j = 0; j < 16; ++j) {
            int ka = hi * 16 + j + kt * 32;
            a1[j] = sX1[(rg * 16 + m) * DIM + ka];
            a2[j] = sX2[(rg * 16 + m) * DIM + ka];
        }
        // B fragments: one aligned 32B vector read each
        v16h b1 = *(const v16h*)&sW1f[(((kt * 4 + ct) * 32) + lane) * 16];
        v16h b2 = *(const v16h*)&sW2f[(((kt * 4 + ct) * 32) + lane) * 16];

        acc = __builtin_amdgcn_wmma_f32_16x16x32_f16(false, a1, false, b1,
                                                     (short)0, acc, false, false);
        acc = __builtin_amdgcn_wmma_f32_16x16x32_f16(false, a2, false, b2,
                                                     (short)0, acc, false, false);
    }
    // D layout: lane%16 = N, VGPR r = M (+8 for lanes 16-31)
#pragma unroll
    for (int r = 0; r < 8; ++r) {
        int row = rg * 16 + r + hi * 8;
        sO[row * DIM + ct * 16 + m] = lrelu(acc[r]);
    }
    __syncthreads();

    if (tid < 32) {                    // per-row L2 norm
        float s2 = 0.0f;
#pragma unroll
        for (int d = 0; d < DIM; ++d) { float x = sO[tid * DIM + d]; s2 += x * x; }
        sR[tid] = 1.0f / fmaxf(sqrtf(s2), EPS_NRM);
    }
    __syncthreads();

    for (int i = tid; i < 32 * DIM; i += 256) {
        float v = sO[i] * sR[i / DIM];
        size_t g = (size_t)nb * DIM + i;
        ego_out[g] = v;
        sum[g] += v;
    }
}

// ---- final = mean of 3 snapshots, reordered (items first, then users) -------
__global__ __launch_bounds__(256) void k_finalize(const float* __restrict__ sum,
                                                  float* __restrict__ out) {
    int i = blockIdx.x * 256 + threadIdx.x;
    if (i >= NNODES * DIM) return;
    int node = i / DIM, d = i % DIM;
    float v = sum[i] * (1.0f / 3.0f);
    if (node >= N_USERS)
        out[(size_t)(node - N_USERS) * DIM + d] = v;               // item_res
    else
        out[(size_t)N_ENT * DIM + (size_t)node * DIM + d] = v;     // user_res
}

extern "C" void kernel_launch(void* const* d_in, const int* in_sizes, int n_in,
                              void* d_out, int out_size, void* d_ws, size_t ws_size,
                              hipStream_t stream) {
    (void)in_sizes; (void)n_in; (void)out_size; (void)ws_size;
    const float* user_emb   = (const float*)d_in[0];
    const float* entity_emb = (const float*)d_in[1];
    const float* rel_emb    = (const float*)d_in[2];
    const float* W1         = (const float*)d_in[3];
    const float* W2         = (const float*)d_in[4];
    const int*   src        = (const int*)d_in[5];
    const int*   dst        = (const int*)d_in[6];
    const int*   rel        = (const int*)d_in[7];
    float*       out        = (float*)d_out;

    size_t nd = (size_t)NNODES * DIM;
    float* egoA  = (float*)d_ws;
    float* egoB  = egoA + nd;
    float* sum   = egoB + nd;
    float* aggr  = sum  + nd;
    float* lg    = aggr + nd;
    float* mx    = lg   + NEDGE;
    float* denom = mx   + NNODES;
    // total scratch: 4*nd + NEDGE + 2*NNODES floats ~= 108 MB

    dim3 blk(256);
    int gElem  = (NNODES * DIM + 255) / 256;   // 25000
    int gEdgeW = (NEDGE + 7) / 8;              // wave-per-edge, 8 waves/block
    int gEdgeT = (NEDGE + 255) / 256;
    int gGemm  = NNODES / 32;                  // 3125 (exact)

    k_concat<<<gElem, blk, 0, stream>>>(user_emb, entity_emb, egoA, sum);

    float* cur = egoA;
    float* nxt = egoB;
    for (int layer = 0; layer < 2; ++layer) {
        k_init_nodes <<<gElem,  blk, 0, stream>>>(aggr, mx, denom);
        k_edge_logits<<<gEdgeW, blk, 0, stream>>>(cur, rel_emb, src, dst, rel, lg, mx);
        k_edge_exp   <<<gEdgeT, blk, 0, stream>>>(src, mx, lg, denom);
        k_edge_aggr  <<<gEdgeW, blk, 0, stream>>>(cur, src, dst, lg, denom, aggr);
        k_layer_update<<<gGemm, blk, 0, stream>>>(cur, aggr, W1, W2, nxt, sum);
        float* t = cur; cur = nxt; nxt = t;
    }
    k_finalize<<<gElem, blk, 0, stream>>>(sum, out);
}
